// Inv_block_19670950216069
// MI455X (gfx1250) — matrix operand order repaired and checked by reading, compiled, not run on hardware
//
#include <hip/hip_runtime.h>
#include <hip/hip_bf16.h>
#include <math.h>

// ---------------------------------------------------------------------------
// Problem constants (from the reference)
// ---------------------------------------------------------------------------
#define BATCH 2
#define CHN   128
#define IMGH  128
#define IMGW  128
#define HWPX  (IMGH*IMGW)   // 16384
#define KSZ   7
#define GRP   4
#define GCH   16

typedef __attribute__((ext_vector_type(16))) _Float16 v16h;
typedef __attribute__((ext_vector_type(8)))  float    v8f;

__device__ __forceinline__ v8f wmma_f16(v16h a, v16h b, v8f c) {
  return __builtin_amdgcn_wmma_f32_16x16x32_f16(
      false, a, false, b, (short)0, c, false, false);
}

__device__ __forceinline__ int clampi(int v, int lo, int hi) {
  v = (v < lo) ? lo : v;
  return (v > hi) ? hi : v;
}

// ---------------------------------------------------------------------------
// Fused 1x1-conv GEMM over pixels: out[m,p] = sum_c W[m,c]*X[c,p]
// Block = 256 thr (8 waves), 128 consecutive pixels, one 16-row m-tile.
// A and B staged into LDS in WMMA fragment order -> one aligned 32B LDS
// vector read per fragment per lane.  Next tile prefetched via
// global_prefetch_b8 while WMMAs run.
//   EPI: 0=PReLU->f16, 1=ReLU->f16, 2=bias->f16, 3=final f32 combine
// grid = (HWPX/128, COUTP/16, BATCH)
// ---------------------------------------------------------------------------
template<int CIN, int COUTP, int COUTR, bool INF16, int EPI>
__global__ __launch_bounds__(256)
void gemm1x1_kernel(const void* __restrict__ xin,
                    const float* __restrict__ wgt,      // [COUTR, CIN] f32
                    const float* __restrict__ bias,     // [COUTR] f32
                    const float* __restrict__ alpha_p,  // PReLU scalar or null
                    _Float16* __restrict__ out16,
                    float* __restrict__ outf,
                    const _Float16* __restrict__ x3,
                    const float* __restrict__ attn)
{
  constexpr int KB = CIN / 32;
  __shared__ __align__(32) _Float16 walds[KB * 512];  // A frags: [lane][kb][16]
  __shared__ __align__(32) _Float16 blds[8 * 512];    // B frags: [wave][lane][16]

  const int b   = blockIdx.z;
  const int m0  = blockIdx.y * 16;
  const int tid = threadIdx.x;

  // ---- stage A fragments (f32 weights -> f16, fragment order) ----
  for (int i = tid; i < 16 * CIN; i += 256) {
    int r = i / CIN, c = i % CIN;
    int kb = c >> 5, k = c & 31;
    int s = k >> 4;
    int h = k & 15;
    int L = (s << 4) | r;
    int gr = m0 + r;
    int grc = (gr < COUTR) ? gr : (COUTR - 1);
    float wv = wgt[grc * CIN + c];
    walds[(L * KB + kb) * 16 + h] = (gr < COUTR) ? (_Float16)wv : (_Float16)0.0f;
  }
  __syncthreads();

  const int lane = tid & 31;
  const int wave = tid >> 5;
  const int l15  = lane & 15;
  const int s    = lane >> 4;

  v16h afrag[KB];
#pragma unroll
  for (int kb = 0; kb < KB; ++kb)
    afrag[kb] = *reinterpret_cast<const v16h*>(&walds[(lane * KB + kb) * 16]);

  const int pblk = blockIdx.x * 128;
  const int p    = pblk + wave * 16 + l15;

  const int krow = tid >> 3;   // 0..31  (K row inside block)
  const int seg  = tid & 7;    // 0..7   (16-pixel segment == destination wave)
  const int s2   = krow >> 4;
  const int h2   = krow & 15;

  v8f acc = {};
#pragma unroll
  for (int kb = 0; kb < KB; ++kb) {
    __syncthreads();
    // ---- cooperative coalesced stage of 32(K) x 128(N) tile, frag order ----
    {
      int c = kb * 32 + krow;
      size_t srcbase = (size_t)(b * CIN + c) * HWPX + pblk + seg * 16;
      _Float16 vals[16];
      if (INF16) {
        const _Float16* sp = (const _Float16*)xin + srcbase;
        v16h vv = *reinterpret_cast<const v16h*>(sp);
        __builtin_prefetch(sp + 128, 0, 0);   // next pixel block's row
#pragma unroll
        for (int j = 0; j < 16; ++j) vals[j] = vv[j];
      } else {
        const float* spf = (const float*)xin + srcbase;
        const float4* s4 = reinterpret_cast<const float4*>(spf);
        float4 q0 = s4[0], q1 = s4[1], q2 = s4[2], q3 = s4[3];
        __builtin_prefetch(spf + 128, 0, 0);  // next pixel block's row
        vals[0]=(_Float16)q0.x; vals[1]=(_Float16)q0.y; vals[2]=(_Float16)q0.z; vals[3]=(_Float16)q0.w;
        vals[4]=(_Float16)q1.x; vals[5]=(_Float16)q1.y; vals[6]=(_Float16)q1.z; vals[7]=(_Float16)q1.w;
        vals[8]=(_Float16)q2.x; vals[9]=(_Float16)q2.y; vals[10]=(_Float16)q2.z; vals[11]=(_Float16)q2.w;
        vals[12]=(_Float16)q3.x; vals[13]=(_Float16)q3.y; vals[14]=(_Float16)q3.z; vals[15]=(_Float16)q3.w;
      }
#pragma unroll
      for (int j = 0; j < 16; ++j)
        blds[seg * 512 + (((s2 << 4) | j) << 4) + h2] = vals[j];
    }
    __syncthreads();
    v16h bf = *reinterpret_cast<const v16h*>(&blds[wave * 512 + lane * 16]);
    acc = wmma_f16(afrag[kb], bf, acc);
  }

  const float alpha = (EPI == 0) ? alpha_p[0] : 0.0f;

#pragma unroll
  for (int v = 0; v < 8; ++v) {
    int m  = m0 + v + 8 * s;
    int mc = (m < COUTR) ? m : (COUTR - 1);
    float bterm = bias[mc];
    float val = acc[v] + ((m < COUTR) ? bterm : 0.0f);
    if (EPI == 0) {
      val = (val >= 0.0f) ? val : alpha * val;
      out16[(size_t)(b * COUTP + m) * HWPX + p] = (_Float16)val;
    } else if (EPI == 1) {
      val = (val > 0.0f) ? val : 0.0f;
      out16[(size_t)(b * COUTP + m) * HWPX + p] = (_Float16)val;
    } else if (EPI == 2) {
      out16[(size_t)(b * COUTP + m) * HWPX + p] = (_Float16)val;
    } else {
      float av  = attn[(size_t)b * HWPX + p];
      float x3v = (float)x3[(size_t)(b * 128 + m) * HWPX + p];
      outf[((size_t)(b * 128 + m) << 14) + p] = x3v * av + val;
    }
  }
}

// ---------------------------------------------------------------------------
// Involution multiply-reduce + PReLU(a2).
// Block = 256 thr = 2 image rows.  Per group, a zero-filled 8x134x16ch halo
// tile of x1 is staged into LDS with coalesced clamped loads; the 16x49 MAC
// inner loop then reads LDS with no bounds logic at all.
// grid = (B*HW/256)
// ---------------------------------------------------------------------------
__global__ __launch_bounds__(256)
void involution_kernel(const _Float16* __restrict__ x1,   // [B,64,HW]
                       const _Float16* __restrict__ kw,   // [B,208,HW]
                       const float* __restrict__ a2_p,
                       _Float16* __restrict__ x2out)      // [B,64,HW]
{
  __shared__ _Float16 hlds[8 * 134 * GCH];   // [row][col][ch], 34 KB

  const int tid = threadIdx.x;
  const int t = blockIdx.x * 256 + tid;      // 0 .. B*HW-1
  const int b = t >> 14;
  const int p = t & (HWPX - 1);
  const int hy    = p >> 7;
  const int hyloc = hy & 1;                  // row within the block's pair
  const int hy0   = hy & ~1;                 // first row of the pair
  const int wx    = p & (IMGW - 1);
  const float a2 = a2_p[0];

  for (int g = 0; g < GRP; ++g) {
    __syncthreads();   // previous group's LDS reads complete
    // ---- stage zero-padded halo: rows hy0-3..hy0+4, cols -3..130, 16 ch ----
    for (int i = tid; i < GCH * 8 * 134; i += 256) {
      int ch  = i / (8 * 134);
      int rem = i % (8 * 134);
      int row = rem / 134;
      int col = rem % 134;
      int yy = hy0 + row - 3, xx = col - 3;
      int yc = clampi(yy, 0, IMGH - 1);
      int xc = clampi(xx, 0, IMGW - 1);
      bool inb = (yy == yc) && (xx == xc);
      _Float16 v = x1[(size_t)(b * 64 + g * GCH + ch) * HWPX + (yc << 7) + xc];
      hlds[(row * 134 + col) * GCH + ch] = inb ? v : (_Float16)0.0f;
    }
    __syncthreads();

    // ---- per-pixel dynamic kernel ----
    float kv[49];
    const _Float16* kwb = kw + (size_t)(b * 208 + g * 49) * HWPX + p;
#pragma unroll
    for (int tt = 0; tt < 49; ++tt)
      kv[tt] = (float)kwb[(size_t)tt * HWPX];

    // ---- 16 channels x 49 taps, pure LDS load + FMA ----
    for (int ch = 0; ch < GCH; ++ch) {
      float acc = 0.0f;
      int tt = 0;
#pragma unroll
      for (int i = 0; i < KSZ; ++i) {
#pragma unroll
        for (int j = 0; j < KSZ; ++j, ++tt)
          acc += kv[tt] * (float)hlds[((hyloc + i) * 134 + wx + j) * GCH + ch];
      }
      float r = (acc >= 0.0f) ? acc : a2 * acc;
      x2out[(size_t)(b * 64 + g * GCH + ch) * HWPX + p] = (_Float16)r;
    }
  }
}

// ---------------------------------------------------------------------------
// ZPool: per-pixel max & mean over f's 32 channels
// ---------------------------------------------------------------------------
__global__ __launch_bounds__(256)
void zpool_kernel(const _Float16* __restrict__ f, float* __restrict__ zp)
{
  int t = blockIdx.x * 256 + threadIdx.x;
  int b = t >> 14;
  int p = t & (HWPX - 1);
  float mx = -3.0e38f, sum = 0.0f;
#pragma unroll
  for (int c = 0; c < 32; ++c) {
    float v = (float)f[(size_t)(b * 32 + c) * HWPX + p];
    mx = fmaxf(mx, v);
    sum += v;
  }
  zp[(size_t)(b * 2 + 0) * HWPX + p] = mx;
  zp[(size_t)(b * 2 + 1) * HWPX + p] = sum * (1.0f / 32.0f);
}

// ---------------------------------------------------------------------------
// 7x7 attention conv (2->1, pad 3) + sigmoid. Branch-free clamp + mask.
// ---------------------------------------------------------------------------
__global__ __launch_bounds__(256)
void attn_kernel(const float* __restrict__ zp, const float* __restrict__ wa,
                 const float* __restrict__ ba, float* __restrict__ attn)
{
  int t = blockIdx.x * 256 + threadIdx.x;
  int b = t >> 14;
  int p = t & (HWPX - 1);
  int hy = p >> 7, wx = p & (IMGW - 1);
  float acc = ba[0];
  for (int c = 0; c < 2; ++c) {
    const float* zb = zp + (size_t)(b * 2 + c) * HWPX;
#pragma unroll
    for (int i = 0; i < KSZ; ++i) {
      int yy = hy + i - 3;
      int yc = clampi(yy, 0, IMGH - 1);
#pragma unroll
      for (int j = 0; j < KSZ; ++j) {
        int xx = wx + j - 3;
        int xc = clampi(xx, 0, IMGW - 1);
        float msk = ((yy == yc) && (xx == xc)) ? 1.0f : 0.0f;
        acc += msk * wa[(c * KSZ + i) * KSZ + j] * zb[(yc << 7) + xc];
      }
    }
  }
  attn[(size_t)b * HWPX + p] = 1.0f / (1.0f + expf(-acc));
}

// ---------------------------------------------------------------------------
// psec 3x3 conv (32->32, pad 1) as implicit GEMM.
// Block = one image row (128 px). Zero-filled halo tile [3][130][32ch] in LDS
// with channels contiguous, so each tap's B fragment is one 32B LDS vec load.
// grid = (128, 2, BATCH)
// ---------------------------------------------------------------------------
__global__ __launch_bounds__(256)
void psec3x3_kernel(const _Float16* __restrict__ f,     // [B,32,HW]
                    const float* __restrict__ wp1,      // [32,32,3,3]
                    const float* __restrict__ bp1,
                    const float* __restrict__ ap_p,
                    _Float16* __restrict__ tout)        // [B,32,HW]
{
  __shared__ __align__(32) _Float16 walds[32 * 9 * 16];     // [lane][tap][16]
  __shared__ __align__(64) _Float16 hlds[3 * 130 * 32];     // [row][col][ch]

  const int b   = blockIdx.z;
  const int m0  = blockIdx.y * 16;
  const int tid = threadIdx.x;
  const int hy  = blockIdx.x;          // one image row per block

  // A fragments: wp1 (O,I,3,3) f32 -> f16 in fragment order
  for (int i = tid; i < 16 * 288; i += 256) {
    int r = i / 288, rem = i % 288, tap = rem >> 5, k = rem & 31;
    int s = k >> 4;
    int h = k & 15;
    int L = (s << 4) | r;
    walds[(L * 9 + tap) * 16 + h] = (_Float16)wp1[((m0 + r) * 32 + k) * 9 + tap];
  }

  // Halo staging: rows hy-1..hy+1, cols -1..128, all 32 channels.
  for (int i = tid; i < 32 * 390; i += 256) {
    int ch   = i / 390;
    int cell = i % 390;          // 0..389
    int row  = cell / 130;       // 0..2
    int col  = cell % 130;       // 0..129
    int yy = hy + row - 1, xx = col - 1;
    int yc = clampi(yy, 0, IMGH - 1);
    int xc = clampi(xx, 0, IMGW - 1);
    bool inb = (yy == yc) && (xx == xc);
    _Float16 v = f[(size_t)(b * 32 + ch) * HWPX + (yc << 7) + xc]; // clamped
    hlds[(row * 130 + col) * 32 + ch] = inb ? v : (_Float16)0.0f;
  }
  __syncthreads();

  const int lane = tid & 31;
  const int wave = tid >> 5;
  const int l15  = lane & 15;
  const int s    = lane >> 4;

  v16h afrag[9];
#pragma unroll
  for (int tap = 0; tap < 9; ++tap)
    afrag[tap] = *reinterpret_cast<const v16h*>(&walds[(lane * 9 + tap) * 16]);

  const int col = wave * 16 + l15;     // 0..127
  v8f acc = {};
#pragma unroll
  for (int tap = 0; tap < 9; ++tap) {
    int dy = tap / 3, dx = tap % 3;    // 0..2 (halo-relative)
    v16h bf = *reinterpret_cast<const v16h*>(
        &hlds[(dy * 130 + col + dx) * 32 + (s << 4)]);
    acc = wmma_f16(afrag[tap], bf, acc);
  }

  const float ap = ap_p[0];
  const int p = (hy << 7) + col;
#pragma unroll
  for (int v = 0; v < 8; ++v) {
    int m = m0 + v + 8 * s;
    float val = acc[v] + bp1[m];
    val = (val >= 0.0f) ? val : ap * val;
    tout[(size_t)(b * 32 + m) * HWPX + p] = (_Float16)val;
  }
}

// ---------------------------------------------------------------------------
// Host-side orchestration
// ---------------------------------------------------------------------------
extern "C" void kernel_launch(void* const* d_in, const int* in_sizes, int n_in,
                              void* d_out, int out_size, void* d_ws, size_t ws_size,
                              hipStream_t stream) {
  (void)in_sizes; (void)n_in; (void)out_size; (void)ws_size;

  const float* x   = (const float*)d_in[0];
  const float* w1  = (const float*)d_in[1];
  const float* b1  = (const float*)d_in[2];
  const float* a1  = (const float*)d_in[3];
  const float* wr  = (const float*)d_in[4];
  const float* br  = (const float*)d_in[5];
  const float* wsp = (const float*)d_in[6];
  const float* bs  = (const float*)d_in[7];
  const float* a2  = (const float*)d_in[8];
  const float* w2  = (const float*)d_in[9];
  const float* b2  = (const float*)d_in[10];
  const float* wa  = (const float*)d_in[11];
  const float* ba  = (const float*)d_in[12];
  const float* wp1 = (const float*)d_in[13];
  const float* bp1 = (const float*)d_in[14];
  const float* ap  = (const float*)d_in[15];
  const float* wp2 = (const float*)d_in[16];
  const float* bp2 = (const float*)d_in[17];
  float* out = (float*)d_out;

  // Workspace layout (bytes)
  char* wsb = (char*)d_ws;
  _Float16* x1w = (_Float16*)(wsb + 0);              // [2, 64,HW] f16  (4 MB)
  _Float16* fw  = (_Float16*)(wsb + 4194304);        // [2, 32,HW] f16  (2 MB)
  _Float16* kww = (_Float16*)(wsb + 6291456);        // [2,208,HW] f16  (13 MB)
  _Float16* x2w = (_Float16*)(wsb + 19922944);       // [2, 64,HW] f16  (4 MB)
  _Float16* x3w = (_Float16*)(wsb + 24117248);       // [2,128,HW] f16  (8 MB)
  float*    zpw = (float*)   (wsb + 32505856);       // [2,  2,HW] f32
  float*    atw = (float*)   (wsb + 32768000);       // [2,    HW] f32
  _Float16* tw  = (_Float16*)(wsb + 32899072);       // [2, 32,HW] f16  (2 MB)

  const dim3 blk(256);
  const int PX_BLKS = HWPX / 128;                    // 128

  // 1) conv1 (128->64) + PReLU(a1) -> x1 (f16)
  gemm1x1_kernel<128, 64, 64, false, 0>
      <<<dim3(PX_BLKS, 4, BATCH), blk, 0, stream>>>(x, w1, b1, a1, x1w,
                                                    nullptr, nullptr, nullptr);
  // 2) reduce (64->32) + ReLU -> f (f16)
  gemm1x1_kernel<64, 32, 32, true, 1>
      <<<dim3(PX_BLKS, 2, BATCH), blk, 0, stream>>>(x1w, wr, br, nullptr, fw,
                                                    nullptr, nullptr, nullptr);
  // 3) span (32->196, padded 208) -> kw (f16)
  gemm1x1_kernel<32, 208, 196, true, 2>
      <<<dim3(PX_BLKS, 13, BATCH), blk, 0, stream>>>(fw, wsp, bs, nullptr, kww,
                                                     nullptr, nullptr, nullptr);
  // 4) involution multiply-reduce + PReLU(a2) -> x2 (f16)
  involution_kernel<<<dim3(BATCH * HWPX / 256), blk, 0, stream>>>(x1w, kww, a2, x2w);
  // 5) conv2 (64->128) -> x3 (f16)
  gemm1x1_kernel<64, 128, 128, true, 2>
      <<<dim3(PX_BLKS, 8, BATCH), blk, 0, stream>>>(x2w, w2, b2, nullptr, x3w,
                                                    nullptr, nullptr, nullptr);
  // 6) ZPool over f -> zp
  zpool_kernel<<<dim3(BATCH * HWPX / 256), blk, 0, stream>>>(fw, zpw);
  // 7) 7x7 attn conv + sigmoid -> attn
  attn_kernel<<<dim3(BATCH * HWPX / 256), blk, 0, stream>>>(zpw, wa, ba, atw);
  // 8) psec 3x3 (32->32) + PReLU(ap) -> t (f16)
  psec3x3_kernel<<<dim3(PX_BLKS, 2, BATCH), blk, 0, stream>>>(fw, wp1, bp1, ap, tw);
  // 9) psec 1x1 (32->128) + combine: out = x3*attn + ps (f32)
  gemm1x1_kernel<32, 128, 128, true, 3>
      <<<dim3(PX_BLKS, 8, BATCH), blk, 0, stream>>>(tw, wp2, bp2, nullptr, nullptr,
                                                    out, x3w, atw);
}
